// GraphSAGE_27084063769012
// MI455X (gfx1250) — matrix operand order repaired and checked by reading, not compiled, not run on hardware
//
#include <hip/hip_runtime.h>
#include <hip/hip_bf16.h>

typedef __attribute__((ext_vector_type(16))) __bf16 v16bf;
typedef __attribute__((ext_vector_type(8)))  float  v8f;

// ---------------------------------------------------------------------------
// Utility: zero a float buffer (grid-stride)
// ---------------------------------------------------------------------------
__global__ void sage_zero_kernel(float* __restrict__ p, long long n) {
    long long i = (long long)blockIdx.x * blockDim.x + threadIdx.x;
    long long stride = (long long)gridDim.x * blockDim.x;
    for (; i < n; i += stride) p[i] = 0.0f;
}

// ---------------------------------------------------------------------------
// Pre-pack weights: Wcat[n][0:128] = Wl[n][:], Wcat[n][128:256] = Wr[n][:],
// converted to bf16 once per call so the GEMM inner loop does zero converts.
// ---------------------------------------------------------------------------
__global__ void sage_wcat_kernel(const float* __restrict__ Wl,
                                 const float* __restrict__ Wr,
                                 __bf16* __restrict__ Wcat, int N) {
    int i = blockIdx.x * blockDim.x + threadIdx.x;
    if (i >= N * 256) return;
    int n = i >> 8;
    int k = i & 255;
    float v = (k < 128) ? Wl[n * 128 + k] : Wr[n * 128 + (k - 128)];
    Wcat[i] = (__bf16)v;
}

// ---------------------------------------------------------------------------
// Edge scatter: agg[dst] += feat[src] (128 channels), optional degree count.
// 32 threads per edge, 4 channels (float4) per thread -> global_atomic_add_f32
// ---------------------------------------------------------------------------
__global__ void sage_scatter_kernel(const float* __restrict__ feat,
                                    const int* __restrict__ src,
                                    const int* __restrict__ dst,
                                    float* __restrict__ agg,
                                    float* __restrict__ deg,
                                    int nEdges, int countDeg) {
    long long tid = (long long)blockIdx.x * blockDim.x + threadIdx.x;
    long long total = (long long)nEdges * 32;
    if (tid >= total) return;
    int e  = (int)(tid >> 5);
    int cg = (int)(tid & 31);
    int s = src[e];
    int d = dst[e];
    float4 v = *(const float4*)(feat + (size_t)s * 128 + cg * 4);
    float* o = agg + (size_t)d * 128 + cg * 4;
    atomicAdd(o + 0, v.x);
    atomicAdd(o + 1, v.y);
    atomicAdd(o + 2, v.z);
    atomicAdd(o + 3, v.w);
    if (countDeg && cg == 0) atomicAdd(deg + d, 1.0f);
}

// ---------------------------------------------------------------------------
// deg -> deg_inv in place (matches reference: where(deg>0, 1/deg, 0))
// ---------------------------------------------------------------------------
__global__ void sage_deginv_kernel(float* __restrict__ deg, int n) {
    int i = blockIdx.x * blockDim.x + threadIdx.x;
    if (i < n) {
        float d = deg[i];
        deg[i] = (d > 0.0f) ? (1.0f / d) : 0.0f;
    }
}

// ---------------------------------------------------------------------------
// f32x16 -> bf16x16 fragment conversion (with scale) for the A side
// ---------------------------------------------------------------------------
__device__ inline v16bf cvt16(float4 a, float4 b, float4 c, float4 d, float s) {
    v16bf r;
    r[0]  = (__bf16)(a.x * s); r[1]  = (__bf16)(a.y * s);
    r[2]  = (__bf16)(a.z * s); r[3]  = (__bf16)(a.w * s);
    r[4]  = (__bf16)(b.x * s); r[5]  = (__bf16)(b.y * s);
    r[6]  = (__bf16)(b.z * s); r[7]  = (__bf16)(b.w * s);
    r[8]  = (__bf16)(c.x * s); r[9]  = (__bf16)(c.y * s);
    r[10] = (__bf16)(c.z * s); r[11] = (__bf16)(c.w * s);
    r[12] = (__bf16)(d.x * s); r[13] = (__bf16)(d.y * s);
    r[14] = (__bf16)(d.z * s); r[15] = (__bf16)(d.w * s);
    return r;
}

// ---------------------------------------------------------------------------
// Fused SAGE linear: out[M,N] = act( (Aagg*dinv) @ Wl^T + Aself @ Wr^T + b )
// Concatenated K = 256 (128 agg + 128 self), consumed as 8 chunks of K=32 via
// v_wmma_f32_16x16x32_bf16.  Block = 256 threads = 8 waves; each wave owns a
// 16-row strip (block covers 128 rows) and sweeps N in 16-wide tiles.
//
// A fragment (16x32 bf16, wave32):  m = lane&15, kbase = (lane>>4)*16,
//   elems j=0..15 -> k=kbase+j
// B fragment (32x16 bf16):          n = lane&15, k = (lane>>4)*16 + j
//   -> 16 consecutive bf16 from pre-packed Wcat: one aligned 32B vector load.
// C/D (16x16 f32): VGPR i -> M=i (lanes 0-15) / M=i+8 (lanes 16-31), N=lane&15
// ---------------------------------------------------------------------------
__global__ __launch_bounds__(256) void sage_gemm_kernel(
    const float*  __restrict__ Aagg,   // [M,128] unscaled aggregate
    const float*  __restrict__ dinv,   // [M] 1/deg
    const float*  __restrict__ Aself,  // [M,128] self features
    const __bf16* __restrict__ Wcat,   // [N,256] pre-packed bf16 [Wl|Wr]
    const float*  __restrict__ bias,   // [N]
    float* __restrict__ out,           // [M,N]
    int M, int N, int doRelu) {
    const int lane = threadIdx.x & 31;
    const int wave = threadIdx.x >> 5;
    const int row0 = blockIdx.x * 128 + wave * 16;

    const int mrow = row0 + (lane & 15);
    const int m    = (mrow < M) ? mrow : (M - 1);   // clamp loads; mask stores
    const int kbase = (lane >> 4) * 16;             // A fragment k-base
    const int khB   = (lane >> 4) * 16;             // B fragment k-base
    const int ncol  = lane & 15;
    const float sc  = dinv[m];
    const bool fullStrip = (row0 + 16 <= M);        // wave-uniform tail check

    // Hoist all 8 A-chunk fragments for this wave's 16-row strip.
    v16bf afrag[8];
#pragma unroll
    for (int kc = 0; kc < 8; ++kc) {
        const float* srcp = (kc < 4)
            ? (Aagg  + (size_t)m * 128 + kc * 32)
            : (Aself + (size_t)m * 128 + (kc - 4) * 32);
        const float s = (kc < 4) ? sc : 1.0f;
        float4 r0 = *(const float4*)(srcp + kbase);
        float4 r1 = *(const float4*)(srcp + kbase + 4);
        float4 r2 = *(const float4*)(srcp + kbase + 8);
        float4 r3 = *(const float4*)(srcp + kbase + 12);
        afrag[kc] = cvt16(r0, r1, r2, r3, s);
    }

    const int nTiles = N >> 4;
    for (int nt = 0; nt < nTiles; ++nt) {
        const int n = nt * 16 + ncol;
        const __bf16* wrow = Wcat + (size_t)n * 256 + khB;
        v8f acc = {};
#pragma unroll
        for (int kc = 0; kc < 8; ++kc) {
            // element offset n*256 + kc*32 + khB is a multiple of 16
            // -> byte offset multiple of 32: aligned v16bf load (2x b128)
            v16bf bfrag = *(const v16bf*)(wrow + kc * 32);
            acc = __builtin_amdgcn_wmma_f32_16x16x32_bf16(
                false, afrag[kc], false, bfrag, (short)0, acc, false, false);
        }
        const float bv = bias[n];
        const int moff = row0 + ((lane >> 4) * 8);
        float* orow = out + (size_t)moff * N + n;
        if (fullStrip) {
            // Fast path (all but the last block): straight-line stores.
#pragma unroll
            for (int i = 0; i < 8; ++i) {
                float v = acc[i] + bv;
                if (doRelu) v = fmaxf(v, 0.0f);
                orow[(size_t)i * N] = v;
            }
        } else {
#pragma unroll
            for (int i = 0; i < 8; ++i) {
                float v = acc[i] + bv;
                if (doRelu) v = fmaxf(v, 0.0f);
                if (moff + i < M) orow[(size_t)i * N] = v;
            }
        }
    }
}

// ---------------------------------------------------------------------------
// Host-side orchestration
// ---------------------------------------------------------------------------
extern "C" void kernel_launch(void* const* d_in, const int* in_sizes, int n_in,
                              void* d_out, int out_size, void* d_ws, size_t ws_size,
                              hipStream_t stream) {
    const float* x   = (const float*)d_in[0];
    const int*   ei  = (const int*)d_in[1];   // int64 in ref, but JAX x64-off -> int32
    const float* W1l = (const float*)d_in[2];
    const float* W1r = (const float*)d_in[3];
    const float* b1  = (const float*)d_in[4];
    const float* W2l = (const float*)d_in[5];
    const float* W2r = (const float*)d_in[6];
    const float* b2  = (const float*)d_in[7];
    float* out = (float*)d_out;

    const int M = in_sizes[0] / 128;   // 100000 nodes
    const int E = in_sizes[1] / 2;     // 1600000 edges
    const int* src = ei;
    const int* dst = ei + E;

    // Workspace layout: deg | agg[M,128] | h[M,128] | Wcat1[128,256] | Wcat2[64,256]
    char* ws = (char*)d_ws;
    size_t off = 0;
    float*  deg   = (float*)(ws + off);  off += (((size_t)M * 4) + 255) & ~(size_t)255;
    float*  agg   = (float*)(ws + off);  off += (((size_t)M * 128 * 4) + 255) & ~(size_t)255;
    float*  h     = (float*)(ws + off);  off += (((size_t)M * 128 * 4) + 255) & ~(size_t)255;
    __bf16* Wcat1 = (__bf16*)(ws + off); off += (((size_t)128 * 256 * 2) + 255) & ~(size_t)255;
    __bf16* Wcat2 = (__bf16*)(ws + off);

    const long long aggN = (long long)M * 128;
    const long long scatterThreads = (long long)E * 32;
    const int scatterBlocks = (int)((scatterThreads + 255) / 256);
    const int gemmBlocks = (M + 127) / 128;

    // Weight pre-pack (tiny, once per call)
    sage_wcat_kernel<<<(128 * 256 + 255) / 256, 256, 0, stream>>>(W1l, W1r, Wcat1, 128);
    sage_wcat_kernel<<<(64 * 256 + 255) / 256, 256, 0, stream>>>(W2l, W2r, Wcat2, 64);

    // ---- Layer 1 ----
    sage_zero_kernel<<<2048, 256, 0, stream>>>(deg, (long long)M);
    sage_zero_kernel<<<4096, 256, 0, stream>>>(agg, aggN);
    sage_scatter_kernel<<<scatterBlocks, 256, 0, stream>>>(x, src, dst, agg, deg, E, 1);
    sage_deginv_kernel<<<(M + 255) / 256, 256, 0, stream>>>(deg, M);
    sage_gemm_kernel<<<gemmBlocks, 256, 0, stream>>>(
        agg, deg, x, Wcat1, b1, h, M, 128, 1);

    // ---- Layer 2 ----
    sage_zero_kernel<<<4096, 256, 0, stream>>>(agg, aggN);
    sage_scatter_kernel<<<scatterBlocks, 256, 0, stream>>>(h, src, dst, agg, deg, E, 0);
    sage_gemm_kernel<<<gemmBlocks, 256, 0, stream>>>(
        agg, deg, h, Wcat2, b2, out, M, 64, 0);
}
